// GAT_76836964925868
// MI455X (gfx1250) — compile-verified, matrix operand order checked by reading, not run on hardware
//
#include <hip/hip_runtime.h>
#include <stdint.h>

// MI455X / gfx1250, wave32. GAT matmuls on v_wmma_f32_16x16x32_bf16 with
// operands pre-swizzled into WMMA fragment order in LDS (2x ds_load_b128 per
// fragment). Uniform full-tile fast path removes per-row exec-mask ladders.
// Attention/softmax/aggregation on VALU (L2-resident gathers), sequential
// GRU+physics on a single wave. Workspace requirement ~200 MB.

#define TSTEPS 32
#define DEG    32

typedef __attribute__((ext_vector_type(16))) __bf16          v16bf;
typedef __attribute__((ext_vector_type(8)))  float           v8f;
typedef __attribute__((ext_vector_type(8)))  unsigned short  v8us;

__device__ __forceinline__ unsigned short f2bf(float f) {
  unsigned u = __float_as_uint(f);
  u += 0x7FFFu + ((u >> 16) & 1u);     // round-to-nearest-even
  return (unsigned short)(u >> 16);
}
__device__ __forceinline__ float bf2f(unsigned short s) {
  return __uint_as_float(((unsigned)s) << 16);
}
__device__ __forceinline__ float sigmf(float x) { return 1.f / (1.f + __expf(-x)); }

// Fragment element mapping (16-bit A/B, 16x16x32):
//   lane half hi = (kk>>3)&1, element half = (kk&16)?1:0, offset = kk&7.
// An 8-aligned run of K maps to one contiguous 8-element (16B) half of one
// lane's v16bf fragment -> fill with single ds_store_b128 per group.

// ---------------------------------------------------------------------------
// Layer-1 matmul: z1[t,n,c] = sum_d h[t,n,d] * W1[c/32, d, c%32], c in [0,128)
// Block = 128 threads (4 waves); each wave owns a 16-node row tile.
// ---------------------------------------------------------------------------
__global__ void k_gat1_mm(const float* __restrict__ h, const float* __restrict__ W1,
                          unsigned short* __restrict__ z1, int nn, int tiles) {
  __shared__ v16bf afrag[4][4][32];   // [row-tile][kstep][lane]  16 KB
  __shared__ v16bf bfrag[8][4][32];   // [col-tile][kstep][lane]  32 KB
  int t = blockIdx.x / tiles;
  int node0 = (blockIdx.x % tiles) * 64;
  int tid = threadIdx.x;
  bool full = (node0 + 64 <= nn);     // uniform per block

  // B fill: 128 cols x 16 K-groups; one 16B LDS store per group.
  for (int g = tid; g < 128 * 16; g += 128) {
    int c  = g >> 4;
    int d0 = (g & 15) * 8;
    int ks = d0 >> 5;
    int kk = d0 & 31;
    int hi = (kk >> 3) & 1;
    int eh = (kk & 16) ? 1 : 0;
    int ln = hi * 16 + (c & 15);
    const float* wp = W1 + (size_t)(c >> 5) * (128 * 32) + (size_t)d0 * 32 + (c & 31);
    v8us p;
#pragma unroll
    for (int j = 0; j < 8; ++j) p[j] = f2bf(wp[(size_t)j * 32]);
    ((v8us*)&bfrag[c >> 4][ks][ln])[eh] = p;
  }
  // A fill: 64 rows x 16 K-groups; 32B coalesced global read -> 16B LDS store.
  if (full) {
    for (int g = tid; g < 64 * 16; g += 128) {
      int r  = g >> 4;
      int d0 = (g & 15) * 8;
      int kk = d0 & 31;
      int ln = ((kk >> 3) & 1) * 16 + (r & 15);
      int eh = (kk & 16) ? 1 : 0;
      const float* hp = h + ((size_t)t * nn + node0 + r) * 128 + d0;
      v8us p;
#pragma unroll
      for (int j = 0; j < 8; ++j) p[j] = f2bf(hp[j]);
      ((v8us*)&afrag[r >> 4][d0 >> 5][ln])[eh] = p;
    }
  } else {
    for (int g = tid; g < 64 * 16; g += 128) {
      int r  = g >> 4;
      int d0 = (g & 15) * 8;
      int kk = d0 & 31;
      int ln = ((kk >> 3) & 1) * 16 + (r & 15);
      int eh = (kk & 16) ? 1 : 0;
      int n  = node0 + r;
      v8us p;
      if (n < nn) {
        const float* hp = h + ((size_t)t * nn + n) * 128 + d0;
#pragma unroll
        for (int j = 0; j < 8; ++j) p[j] = f2bf(hp[j]);
      } else {
#pragma unroll
        for (int j = 0; j < 8; ++j) p[j] = (unsigned short)0;
      }
      ((v8us*)&afrag[r >> 4][d0 >> 5][ln])[eh] = p;
    }
  }
  __syncthreads();

  int lane = tid & 31;
  int rt   = tid >> 5;
  int moff = (lane >> 4) * 8;
  v16bf A0 = afrag[rt][0][lane];
  v16bf A1 = afrag[rt][1][lane];
  v16bf A2 = afrag[rt][2][lane];
  v16bf A3 = afrag[rt][3][lane];

  for (int ct = 0; ct < 8; ++ct) {
    v16bf B0 = bfrag[ct][0][lane];
    v16bf B1 = bfrag[ct][1][lane];
    v16bf B2 = bfrag[ct][2][lane];
    v16bf B3 = bfrag[ct][3][lane];
    v8f acc = {};
    acc = __builtin_amdgcn_wmma_f32_16x16x32_bf16(false, A0, false, B0, (short)0, acc, false, false);
    acc = __builtin_amdgcn_wmma_f32_16x16x32_bf16(false, A1, false, B1, (short)0, acc, false, false);
    acc = __builtin_amdgcn_wmma_f32_16x16x32_bf16(false, A2, false, B2, (short)0, acc, false, false);
    acc = __builtin_amdgcn_wmma_f32_16x16x32_bf16(false, A3, false, B3, (short)0, acc, false, false);

    if (full) {
      unsigned short* zp =
          z1 + ((size_t)t * nn + node0 + rt * 16 + moff) * 128 + ct * 16 + (lane & 15);
#pragma unroll
      for (int v = 0; v < 8; ++v) zp[(size_t)v * 128] = f2bf(acc[v]);
    } else {
#pragma unroll
      for (int v = 0; v < 8; ++v) {
        int n = node0 + rt * 16 + moff + v;
        if (n < nn)
          z1[((size_t)t * nn + n) * 128 + ct * 16 + (lane & 15)] = f2bf(acc[v]);
      }
    }
  }
}

// ---------------------------------------------------------------------------
// Layer-2 matmul: z2[t,n,e] = sum_d h1[t,n,d] * W2[d,e], e in [0,32)
// ---------------------------------------------------------------------------
__global__ void k_gat2_mm(const unsigned short* __restrict__ h1, const float* __restrict__ W2,
                          unsigned short* __restrict__ z2, int nn, int tiles) {
  __shared__ v16bf afrag[4][4][32];   // 16 KB
  __shared__ v16bf bfrag[2][4][32];   //  8 KB
  int t = blockIdx.x / tiles;
  int node0 = (blockIdx.x % tiles) * 64;
  int tid = threadIdx.x;
  bool full = (node0 + 64 <= nn);

  // B fill: 32 cols x 16 K-groups.
  for (int g = tid; g < 32 * 16; g += 128) {
    int c  = g >> 4;
    int d0 = (g & 15) * 8;
    int kk = d0 & 31;
    int ln = ((kk >> 3) & 1) * 16 + (c & 15);
    int eh = (kk & 16) ? 1 : 0;
    const float* wp = W2 + (size_t)d0 * 32 + c;
    v8us p;
#pragma unroll
    for (int j = 0; j < 8; ++j) p[j] = f2bf(wp[(size_t)j * 32]);
    ((v8us*)&bfrag[c >> 4][d0 >> 5][ln])[eh] = p;
  }
  // A fill: source is already bf16 -> 16B global load, 16B LDS store.
  if (full) {
    for (int g = tid; g < 64 * 16; g += 128) {
      int r  = g >> 4;
      int d0 = (g & 15) * 8;
      int kk = d0 & 31;
      int ln = ((kk >> 3) & 1) * 16 + (r & 15);
      int eh = (kk & 16) ? 1 : 0;
      v8us p = *(const v8us*)(h1 + ((size_t)t * nn + node0 + r) * 128 + d0);
      ((v8us*)&afrag[r >> 4][d0 >> 5][ln])[eh] = p;
    }
  } else {
    for (int g = tid; g < 64 * 16; g += 128) {
      int r  = g >> 4;
      int d0 = (g & 15) * 8;
      int kk = d0 & 31;
      int ln = ((kk >> 3) & 1) * 16 + (r & 15);
      int eh = (kk & 16) ? 1 : 0;
      int n  = node0 + r;
      v8us p;
      if (n < nn) {
        p = *(const v8us*)(h1 + ((size_t)t * nn + n) * 128 + d0);
      } else {
#pragma unroll
        for (int j = 0; j < 8; ++j) p[j] = (unsigned short)0;
      }
      ((v8us*)&afrag[r >> 4][d0 >> 5][ln])[eh] = p;
    }
  }
  __syncthreads();

  int lane = tid & 31;
  int rt   = tid >> 5;
  int moff = (lane >> 4) * 8;
  v16bf A0 = afrag[rt][0][lane];
  v16bf A1 = afrag[rt][1][lane];
  v16bf A2 = afrag[rt][2][lane];
  v16bf A3 = afrag[rt][3][lane];

  for (int ct = 0; ct < 2; ++ct) {
    v16bf B0 = bfrag[ct][0][lane];
    v16bf B1 = bfrag[ct][1][lane];
    v16bf B2 = bfrag[ct][2][lane];
    v16bf B3 = bfrag[ct][3][lane];
    v8f acc = {};
    acc = __builtin_amdgcn_wmma_f32_16x16x32_bf16(false, A0, false, B0, (short)0, acc, false, false);
    acc = __builtin_amdgcn_wmma_f32_16x16x32_bf16(false, A1, false, B1, (short)0, acc, false, false);
    acc = __builtin_amdgcn_wmma_f32_16x16x32_bf16(false, A2, false, B2, (short)0, acc, false, false);
    acc = __builtin_amdgcn_wmma_f32_16x16x32_bf16(false, A3, false, B3, (short)0, acc, false, false);

    if (full) {
      unsigned short* zp =
          z2 + ((size_t)t * nn + node0 + rt * 16 + moff) * 32 + ct * 16 + (lane & 15);
#pragma unroll
      for (int v = 0; v < 8; ++v) zp[(size_t)v * 32] = f2bf(acc[v]);
    } else {
#pragma unroll
      for (int v = 0; v < 8; ++v) {
        int n = node0 + rt * 16 + moff + v;
        if (n < nn)
          z2[((size_t)t * nn + n) * 32 + ct * 16 + (lane & 15)] = f2bf(acc[v]);
      }
    }
  }
}

// ---------------------------------------------------------------------------
// Per-node attention scalars: sL[h,n] = z.a_left, sR[h,n] = z.a_right
// One wave per (t,node); lane = feature; butterfly reductions.
// ---------------------------------------------------------------------------
__global__ void k_scalars1(const unsigned short* __restrict__ z1, const float* __restrict__ a1,
                           float* __restrict__ sL, float* __restrict__ sR, int nn) {
  size_t gt = (size_t)blockIdx.x * blockDim.x + threadIdx.x;
  int gw = (int)(gt >> 5);
  int lane = threadIdx.x & 31;
  if (gw >= TSTEPS * nn) return;
  int t = gw / nn, n = gw - t * nn;
  const unsigned short* zr = z1 + ((size_t)t * nn + n) * 128;
  for (int hh = 0; hh < 4; ++hh) {
    float zv = bf2f(zr[hh * 32 + lane]);
    float l = zv * a1[hh * 64 + lane];
    float r = zv * a1[hh * 64 + 32 + lane];
#pragma unroll
    for (int s = 16; s > 0; s >>= 1) {
      l += __shfl_xor(l, s, 32);
      r += __shfl_xor(r, s, 32);
    }
    if (lane == 0) {
      sL[((size_t)t * 4 + hh) * nn + n] = l;
      sR[((size_t)t * 4 + hh) * nn + n] = r;
    }
  }
}

__global__ void k_scalars2(const unsigned short* __restrict__ z2, const float* __restrict__ a2,
                           float* __restrict__ sL, float* __restrict__ sR, int nn) {
  size_t gt = (size_t)blockIdx.x * blockDim.x + threadIdx.x;
  int gw = (int)(gt >> 5);
  int lane = threadIdx.x & 31;
  if (gw >= TSTEPS * nn) return;
  int t = gw / nn, n = gw - t * nn;
  float zv = bf2f(z2[((size_t)t * nn + n) * 32 + lane]);
  float l = zv * a2[lane];
  float r = zv * a2[32 + lane];
#pragma unroll
  for (int s = 16; s > 0; s >>= 1) {
    l += __shfl_xor(l, s, 32);
    r += __shfl_xor(r, s, 32);
  }
  if (lane == 0) {
    sL[(size_t)t * nn + n] = l;
    sR[(size_t)t * nn + n] = r;
  }
}

// ---------------------------------------------------------------------------
// Layer-1 attention + aggregation: wave per (t,node,head); lane = neighbor
// for softmax, lane = output feature for accumulation (coalesced gathers).
// ---------------------------------------------------------------------------
__global__ void k_agg1(const unsigned short* __restrict__ z1, const int* __restrict__ src,
                       const float* __restrict__ sL, const float* __restrict__ sR,
                       unsigned short* __restrict__ h1, int nn) {
  size_t gt = (size_t)blockIdx.x * blockDim.x + threadIdx.x;
  int gw = (int)(gt >> 5);
  int lane = threadIdx.x & 31;
  if (gw >= TSTEPS * nn * 4) return;
  int hh = gw & 3;
  int rem = gw >> 2;
  int n = rem % nn;
  int t = rem / nn;

  int mdx = src[(size_t)n * DEG + lane];
  float e = sL[((size_t)t * 4 + hh) * nn + mdx] + sR[((size_t)t * 4 + hh) * nn + n];
  e = (e > 0.f) ? e : 0.01f * e;                  // leaky_relu(0.01)
  float mx = e;
#pragma unroll
  for (int s = 16; s > 0; s >>= 1) mx = fmaxf(mx, __shfl_xor(mx, s, 32));
  float ex = __expf(e - mx);
  float sum = ex;
#pragma unroll
  for (int s = 16; s > 0; s >>= 1) sum += __shfl_xor(sum, s, 32);
  float alpha = ex / sum;

  const unsigned short* zb = z1 + (size_t)t * nn * 128 + hh * 32 + lane;
  float acc = 0.f;
#pragma unroll 8
  for (int d = 0; d < DEG; ++d) {
    int   md = __shfl(mdx,   d, 32);
    float ad = __shfl(alpha, d, 32);
    acc = fmaf(ad, bf2f(zb[(size_t)md * 128]), acc);
  }
  acc = fmaxf(acc, 0.f);                          // relu
  h1[((size_t)t * nn + n) * 128 + hh * 32 + lane] = f2bf(acc);
}

// ---------------------------------------------------------------------------
// Layer-2 attention + aggregation + node-max pooling (relu>=0 -> uint atomicMax)
// ---------------------------------------------------------------------------
__global__ void k_agg2_max(const unsigned short* __restrict__ z2, const int* __restrict__ src,
                           const float* __restrict__ sL, const float* __restrict__ sR,
                           float* __restrict__ cur, int nn) {
  size_t gt = (size_t)blockIdx.x * blockDim.x + threadIdx.x;
  int gw = (int)(gt >> 5);
  int lane = threadIdx.x & 31;
  if (gw >= TSTEPS * nn) return;
  int t = gw / nn, n = gw - t * nn;

  int mdx = src[(size_t)n * DEG + lane];
  float e = sL[(size_t)t * nn + mdx] + sR[(size_t)t * nn + n];
  e = (e > 0.f) ? e : 0.01f * e;
  float mx = e;
#pragma unroll
  for (int s = 16; s > 0; s >>= 1) mx = fmaxf(mx, __shfl_xor(mx, s, 32));
  float ex = __expf(e - mx);
  float sum = ex;
#pragma unroll
  for (int s = 16; s > 0; s >>= 1) sum += __shfl_xor(sum, s, 32);
  float alpha = ex / sum;

  const unsigned short* zb = z2 + (size_t)t * nn * 32 + lane;
  float acc = 0.f;
#pragma unroll 8
  for (int d = 0; d < DEG; ++d) {
    int   md = __shfl(mdx,   d, 32);
    float ad = __shfl(alpha, d, 32);
    acc = fmaf(ad, bf2f(zb[(size_t)md * 32]), acc);
  }
  acc = fmaxf(acc, 0.f);
  atomicMax((unsigned int*)(cur + t * 32 + lane), __float_as_uint(acc));
}

__global__ void k_init_cur(float* __restrict__ cur) { cur[threadIdx.x] = 0.f; }

// ---------------------------------------------------------------------------
// Sequential GRU + readout + SIR physics. One wave32; lane = hidden unit.
// ---------------------------------------------------------------------------
__global__ void k_gru_seq(const float* __restrict__ cur,
                          const float* __restrict__ I,  const float* __restrict__ R,
                          const float* __restrict__ S,  const float* __restrict__ It,
                          const float* __restrict__ Rt, const float* __restrict__ hx0,
                          const float* __restrict__ W_ih, const float* __restrict__ W_hh,
                          const float* __restrict__ b_ih, const float* __restrict__ b_hh,
                          const float* __restrict__ Wr1,  const float* __restrict__ br1,
                          const float* __restrict__ Wr2,  const float* __restrict__ br2,
                          const float* __restrict__ Nptr, float* __restrict__ out) {
  __shared__ float hx[32];
  __shared__ float nh[34];
  int lane = threadIdx.x;
  float Ns = Nptr[0];
  hx[lane] = hx0[lane];
  __syncthreads();

  for (int t = 0; t < TSTEPS; ++t) {
    float gi0 = b_ih[lane], gi1 = b_ih[32 + lane], gi2 = b_ih[64 + lane];
    float gh0 = b_hh[lane], gh1 = b_hh[32 + lane], gh2 = b_hh[64 + lane];
    for (int k = 0; k < 32; ++k) {
      float c  = cur[t * 32 + k];
      float hk = hx[k];
      gi0 = fmaf(c,  W_ih[(size_t)lane * 32 + k],        gi0);
      gi1 = fmaf(c,  W_ih[(size_t)(32 + lane) * 32 + k], gi1);
      gi2 = fmaf(c,  W_ih[(size_t)(64 + lane) * 32 + k], gi2);
      gh0 = fmaf(hk, W_hh[(size_t)lane * 32 + k],        gh0);
      gh1 = fmaf(hk, W_hh[(size_t)(32 + lane) * 32 + k], gh1);
      gh2 = fmaf(hk, W_hh[(size_t)(64 + lane) * 32 + k], gh2);
    }
    float r  = sigmf(gi0 + gh0);
    float zg = sigmf(gi1 + gh1);
    float ng = tanhf(gi2 + r * gh2);
    float hnew = (1.f - zg) * ng + zg * hx[lane];
    __syncthreads();
    hx[lane] = hnew;
    nh[lane] = hnew;
    if (lane == 0) { nh[32] = It[t]; nh[33] = Rt[t]; }
    __syncthreads();

    float pv = 0.f, av = 0.f;
    if (lane < 10) {
      pv = br1[lane];
      for (int k = 0; k < 34; ++k) pv = fmaf(nh[k], Wr1[lane * 34 + k], pv);
      if ((lane & 1) == 0) out[t * 5 + (lane >> 1)]       = pv;   // nI
      else                 out[160 + t * 5 + (lane >> 1)] = pv;   // nR
    }
    if (lane >= 10 && lane < 12) {
      int j = lane - 10;
      av = br2[j];
      for (int k = 0; k < 34; ++k) av = fmaf(nh[k], Wr2[j * 34 + k], av);
    }
    float ab0 = __shfl(av, 10, 32);
    float ab1 = __shfl(av, 11, 32);
    if (lane == 0) {
      float a_s = sigmf(ab0), b_s = sigmf(ab1);
      float lI = I[t], lR = R[t], lS = S[t];
      float dI = 0.f, dR = 0.f;
      for (int i = 0; i < 5; ++i) {
        if (i > 0) { lI += dI; lR += dR; lS = Ns - lI - lR; }
        dI = a_s * lI * (lS / Ns) - b_s * lI;
        dR = b_s * lI;
        out[320 + t * 5 + i] = dI;   // pI
        out[480 + t * 5 + i] = dR;   // pR
      }
    }
    __syncthreads();
  }
}

// ---------------------------------------------------------------------------
extern "C" void kernel_launch(void* const* d_in, const int* in_sizes, int n_in,
                              void* d_out, int out_size, void* d_ws, size_t ws_size,
                              hipStream_t stream) {
  (void)n_in; (void)out_size; (void)ws_size;
  const float* h    = (const float*)d_in[0];
  const int*   src  = (const int*)d_in[1];
  const float* Np   = (const float*)d_in[2];
  const float* I    = (const float*)d_in[3];
  const float* R    = (const float*)d_in[4];
  const float* S    = (const float*)d_in[5];
  const float* It   = (const float*)d_in[6];
  const float* Rt   = (const float*)d_in[7];
  const float* hx0  = (const float*)d_in[8];
  const float* W1   = (const float*)d_in[9];
  const float* a1   = (const float*)d_in[10];
  const float* W2   = (const float*)d_in[11];
  const float* a2   = (const float*)d_in[12];
  const float* W_ih = (const float*)d_in[13];
  const float* W_hh = (const float*)d_in[14];
  const float* b_ih = (const float*)d_in[15];
  const float* b_hh = (const float*)d_in[16];
  const float* Wr1  = (const float*)d_in[17];
  const float* br1  = (const float*)d_in[18];
  const float* Wr2  = (const float*)d_in[19];
  const float* br2  = (const float*)d_in[20];
  float* out = (float*)d_out;

  const int nn = in_sizes[1] / DEG;           // 10000 nodes
  char* ws = (char*)d_ws;
  size_t off = 0;
  auto alloc = [&](size_t bytes) {
    off = (off + 255) & ~(size_t)255;
    size_t o = off; off += bytes; return o;
  };
  unsigned short* z1  = (unsigned short*)(ws + alloc((size_t)TSTEPS * nn * 128 * 2));
  unsigned short* h1b = (unsigned short*)(ws + alloc((size_t)TSTEPS * nn * 128 * 2));
  unsigned short* z2  = (unsigned short*)(ws + alloc((size_t)TSTEPS * nn * 32 * 2));
  float* sL1 = (float*)(ws + alloc((size_t)TSTEPS * 4 * nn * 4));
  float* sR1 = (float*)(ws + alloc((size_t)TSTEPS * 4 * nn * 4));
  float* sL2 = (float*)(ws + alloc((size_t)TSTEPS * nn * 4));
  float* sR2 = (float*)(ws + alloc((size_t)TSTEPS * nn * 4));
  float* cur = (float*)(ws + alloc((size_t)TSTEPS * 32 * 4));

  int tiles = (nn + 63) / 64;

  k_gat1_mm<<<dim3(TSTEPS * tiles), dim3(128), 0, stream>>>(h, W1, z1, nn, tiles);

  {
    long long thr = (long long)TSTEPS * nn * 32;
    k_scalars1<<<dim3((unsigned)((thr + 255) / 256)), dim3(256), 0, stream>>>(z1, a1, sL1, sR1, nn);
  }
  {
    long long thr = (long long)TSTEPS * nn * 4 * 32;
    k_agg1<<<dim3((unsigned)((thr + 255) / 256)), dim3(256), 0, stream>>>(z1, src, sL1, sR1, h1b, nn);
  }

  k_gat2_mm<<<dim3(TSTEPS * tiles), dim3(128), 0, stream>>>(h1b, W2, z2, nn, tiles);

  {
    long long thr = (long long)TSTEPS * nn * 32;
    k_scalars2<<<dim3((unsigned)((thr + 255) / 256)), dim3(256), 0, stream>>>(z2, a2, sL2, sR2, nn);
  }

  k_init_cur<<<dim3(1), dim3(TSTEPS * 32), 0, stream>>>(cur);

  {
    long long thr = (long long)TSTEPS * nn * 32;
    k_agg2_max<<<dim3((unsigned)((thr + 255) / 256)), dim3(256), 0, stream>>>(z2, src, sL2, sR2, cur, nn);
  }

  k_gru_seq<<<dim3(1), dim3(32), 0, stream>>>(cur, I, R, S, It, Rt, hx0,
                                              W_ih, W_hh, b_ih, b_hh,
                                              Wr1, br1, Wr2, br2, Np, out);
}
